// LocallyGroupedAttn_81449759801420
// MI455X (gfx1250) — matrix operand order repaired and checked by reading, not compiled, not run on hardware
//
#include <hip/hip_runtime.h>

// ---------------------------------------------------------------------------
// LoFTR locally-grouped linear attention, fused per-window kernel for gfx1250.
// One workgroup (256 threads = 8 wave32) handles one 8x8 window (64 tokens,
// C=256). All GEMMs run on v_wmma_f32_16x16x32_bf16 with fp32 accumulation;
// all intermediates live in LDS (aliased phase-to-phase). Each wave computes
// two N-tiles per A-fragment for ILP; transposed outputs are packed into
// 16-byte ds_store_b128.
// ---------------------------------------------------------------------------

typedef __bf16 bf16;
typedef bf16  v16bf __attribute__((ext_vector_type(16)));
typedef float v8f   __attribute__((ext_vector_type(8)));
typedef unsigned short u16x8 __attribute__((ext_vector_type(8)));
typedef unsigned short u16x4 __attribute__((ext_vector_type(4)));

union FragCast { u16x8 h[2]; v16bf v; };
union Pack8 { bf16 b[8]; u16x8 u; };
union Pack4 { bf16 b[4]; u16x4 u; };

// A-matrix 16x32 bf16 fragment (ISA 7.12.2): lane group g = lane>>4 holds
// row (lane&15); K values { 8g..8g+7 } in VGPR0-3 and { 16+8g..+7 } in
// VGPR4-7 -> two contiguous 16-byte runs per lane.
__device__ __forceinline__ v16bf load_frag_A(const bf16* base, int stride,
                                             int m0, int k0, int lane) {
  const int g = lane >> 4, r = lane & 15;
  const bf16* row = base + (m0 + r) * stride + k0;
  FragCast f;
  f.h[0] = *(const u16x8*)(row + 8 * g);
  f.h[1] = *(const u16x8*)(row + 16 + 8 * g);
  return f.v;
}

// B-matrix 32x16 bf16 fragment: lane holds column (lane&15); K values
// { 8g..8g+7 } and { 16+8g..+7 } (g = lane>>4) of Bt[n][k] = W[n][k].
__device__ __forceinline__ v16bf load_frag_B(const bf16* base, int stride,
                                             int n0, int k0, int lane) {
  const int g = lane >> 4, c = lane & 15;
  const bf16* row = base + (n0 + c) * stride + k0;
  FragCast f;
  f.h[0] = *(const u16x8*)(row + 8 * g);
  f.h[1] = *(const u16x8*)(row + 16 + 8 * g);
  return f.v;
}

__device__ __forceinline__ v8f wmma_bf16(v16bf a, v16bf b, v8f c) {
  // (neg_a, A, neg_b, B, c_mod, C, reuse_a, reuse_b)
  return __builtin_amdgcn_wmma_f32_16x16x32_bf16(false, a, false, b,
                                                 (short)0, c, false, false);
}

__device__ __forceinline__ float elu1(float x) {
  return x > 0.0f ? x + 1.0f : __expf(x);
}

// 8-bf16 vector read + accumulate helpers (one ds_load_b128 each)
__device__ __forceinline__ float sum8f(const bf16* p) {
  Pack8 pk; pk.u = *(const u16x8*)p;
  float s = 0.0f;
#pragma unroll
  for (int j = 0; j < 8; ++j) s += (float)pk.b[j];
  return s;
}
__device__ __forceinline__ void sumsq8(const bf16* p, float& s, float& s2) {
  Pack8 pk; pk.u = *(const u16x8*)p;
#pragma unroll
  for (int j = 0; j < 8; ++j) { float v = (float)pk.b[j]; s += v; s2 += v * v; }
}

// ---- constants -------------------------------------------------------------
#define HW    240
#define WS2   64     // tokens per window (L)
#define NWIN  7200   // 8 * 30 * 30

// LDS layout (bytes); buffers aliased across phases.
#define OFF_HCAT 0        // bf16 [64][512]   x(bf16) | LN1(msg)   (64 KB)
#define OFF_QB   65536    // bf16 [64][256]   Q = elu(q)+1         (32 KB)
#define OFF_KTMB 98304    // bf16 [256][64] Kt -> later Mb [64][256]
#define OFF_VTM1 131072   // bf16 [256][64] Vt -> later M1 [64][256]
#define OFF_KVT  163840   // bf16 [256][32]   KV^T per head        (16 KB)
#define OFF_KSUM 180224   // f32  [256]
#define OFF_Z    181248   // f32  [64][8]
#define OFF_RED  183296   // f32  [64][4][2]
#define SMEM_BYTES 185344
#define OFF_H2   65536    // bf16 [64][512] MLP1 output (aliases QB+KTMB)

// fp32 -> bf16 weight pre-conversion into d_ws (read from L2 by all WGs).
__global__ void convert_weights_kernel(const float* __restrict__ Wq,
                                       const float* __restrict__ Wk,
                                       const float* __restrict__ Wv,
                                       const float* __restrict__ Wm,
                                       const float* __restrict__ W1,
                                       const float* __restrict__ W2,
                                       bf16* __restrict__ out) {
  int i = blockIdx.x * 256 + threadIdx.x;
  if (i < 65536)        out[i] = (bf16)Wq[i];
  else if (i < 131072)  out[i] = (bf16)Wk[i - 65536];
  else if (i < 196608)  out[i] = (bf16)Wv[i - 131072];
  else if (i < 262144)  out[i] = (bf16)Wm[i - 196608];
  else if (i < 524288)  out[i] = (bf16)W1[i - 262144];
  else if (i < 655360)  out[i] = (bf16)W2[i - 524288];
}

__global__ __launch_bounds__(256, 1)
void loftr_window_kernel(const float* __restrict__ x,
                         const bf16* __restrict__ wbf,
                         const float* __restrict__ g1,
                         const float* __restrict__ b1,
                         const float* __restrict__ g2,
                         const float* __restrict__ b2,
                         float* __restrict__ out) {
  extern __shared__ float4 smem_f4[];
  char* smem = (char*)smem_f4;
  bf16*  Hcat = (bf16*)(smem + OFF_HCAT);   // [64][512]
  bf16*  Qb   = (bf16*)(smem + OFF_QB);     // [64][256]
  bf16*  KtMb = (bf16*)(smem + OFF_KTMB);   // Kt[256][64] / Mb[64][256]
  bf16*  VtM1 = (bf16*)(smem + OFF_VTM1);   // Vt[256][64] / M1[64][256]
  bf16*  KVt  = (bf16*)(smem + OFF_KVT);    // [256][32]
  float* Ksum = (float*)(smem + OFF_KSUM);  // [256]
  float* Zb   = (float*)(smem + OFF_Z);     // [64][8]
  float* red  = (float*)(smem + OFF_RED);   // [64][4][2]
  bf16*  H2   = (bf16*)(smem + OFF_H2);     // [64][512]

  const int tid  = threadIdx.x;
  const int wave = tid >> 5;
  const int lane = tid & 31;

  const int w   = blockIdx.x;
  const int b   = w / 900;
  const int rem = w - b * 900;
  const int wh  = rem / 30;
  const int ww  = rem - wh * 30;

  const bf16* Wqb = wbf;                 // [256][256]
  const bf16* Wkb = wbf + 65536;
  const bf16* Wvb = wbf + 131072;
  const bf16* Wmb = wbf + 196608;
  const bf16* W1b = wbf + 262144;        // [512][512]
  const bf16* W2b = wbf + 524288;        // [256][512]

  // ---- Phase 0: gather window tile x -> Hcat[:, :256] (bf16) --------------
  {
    const float4* x4 = (const float4*)x;
    for (int e = tid; e < WS2 * 64; e += 256) {     // 64 rows x 64 float4
      int l  = e >> 6;
      int c4 = e & 63;
      int row = wh * 8 + (l >> 3);
      int col = ww * 8 + (l & 7);
      int pix = b * (HW * HW) + row * HW + col;
      float4 v = x4[pix * 64 + c4];
      Pack4 pk;
      pk.b[0] = (bf16)v.x; pk.b[1] = (bf16)v.y;
      pk.b[2] = (bf16)v.z; pk.b[3] = (bf16)v.w;
      *(u16x4*)&Hcat[l * 512 + c4 * 4] = pk.u;
    }
  }
  __syncthreads();

  // ---- Phase 1a: Q = elu(x@Wq^T)+1 (M=64, N=256, K=256) -> Qb row-major ---
  for (int t = wave; t < 32; t += 8) {
    int m = t & 3, np = t >> 2;
    v8f acc0 = {}, acc1 = {};
#pragma unroll
    for (int kk = 0; kk < 8; ++kk) {
      v16bf a  = load_frag_A(Hcat, 512, m * 16, kk * 32, lane);
      v16bf b0 = load_frag_B(Wqb, 256, (np * 2 + 0) * 16, kk * 32, lane);
      v16bf b1 = load_frag_B(Wqb, 256, (np * 2 + 1) * 16, kk * 32, lane);
      acc0 = wmma_bf16(a, b0, acc0);
      acc1 = wmma_bf16(a, b1, acc1);
    }
    int tokb = m * 16 + 8 * (lane >> 4);
    int c0 = (np * 2) * 16 + (lane & 15), c1 = c0 + 16;
#pragma unroll
    for (int j = 0; j < 8; ++j) {
      Qb[(tokb + j) * 256 + c0] = (bf16)elu1(acc0[j]);
      Qb[(tokb + j) * 256 + c1] = (bf16)elu1(acc1[j]);
    }
  }
  // ---- Phase 1b: K = elu(x@Wk^T)+1 -> Kt [256][64] (packed b128 stores) ---
  for (int t = wave; t < 32; t += 8) {
    int m = t & 3, np = t >> 2;
    v8f acc0 = {}, acc1 = {};
#pragma unroll
    for (int kk = 0; kk < 8; ++kk) {
      v16bf a  = load_frag_A(Hcat, 512, m * 16, kk * 32, lane);
      v16bf b0 = load_frag_B(Wkb, 256, (np * 2 + 0) * 16, kk * 32, lane);
      v16bf b1 = load_frag_B(Wkb, 256, (np * 2 + 1) * 16, kk * 32, lane);
      acc0 = wmma_bf16(a, b0, acc0);
      acc1 = wmma_bf16(a, b1, acc1);
    }
    int tokb = m * 16 + 8 * (lane >> 4);
    int c0 = (np * 2) * 16 + (lane & 15), c1 = c0 + 16;
    Pack8 p0, p1;
#pragma unroll
    for (int j = 0; j < 8; ++j) {
      p0.b[j] = (bf16)elu1(acc0[j]);
      p1.b[j] = (bf16)elu1(acc1[j]);
    }
    *(u16x8*)&KtMb[c0 * 64 + tokb] = p0.u;
    *(u16x8*)&KtMb[c1 * 64 + tokb] = p1.u;
  }
  // ---- Phase 1c: V = x@Wv^T -> Vt [256][64] (packed b128 stores) ----------
  for (int t = wave; t < 32; t += 8) {
    int m = t & 3, np = t >> 2;
    v8f acc0 = {}, acc1 = {};
#pragma unroll
    for (int kk = 0; kk < 8; ++kk) {
      v16bf a  = load_frag_A(Hcat, 512, m * 16, kk * 32, lane);
      v16bf b0 = load_frag_B(Wvb, 256, (np * 2 + 0) * 16, kk * 32, lane);
      v16bf b1 = load_frag_B(Wvb, 256, (np * 2 + 1) * 16, kk * 32, lane);
      acc0 = wmma_bf16(a, b0, acc0);
      acc1 = wmma_bf16(a, b1, acc1);
    }
    int tokb = m * 16 + 8 * (lane >> 4);
    int c0 = (np * 2) * 16 + (lane & 15), c1 = c0 + 16;
    Pack8 p0, p1;
#pragma unroll
    for (int j = 0; j < 8; ++j) {
      p0.b[j] = (bf16)acc0[j];
      p1.b[j] = (bf16)acc1[j];
    }
    *(u16x8*)&VtM1[c0 * 64 + tokb] = p0.u;
    *(u16x8*)&VtM1[c1 * 64 + tokb] = p1.u;
  }
  __syncthreads();

  // ---- Phase 2a: Ksum and Z normalizers -----------------------------------
  {
    float s = 0.0f;
#pragma unroll
    for (int l8 = 0; l8 < 8; ++l8) s += sum8f(&KtMb[tid * 64 + l8 * 8]);
    Ksum[tid] = s;
  }
  __syncthreads();
  for (int z = tid; z < WS2 * 8; z += 256) {
    int l = z >> 3, h = z & 7;
    float s = 0.0f;
#pragma unroll
    for (int d8 = 0; d8 < 4; ++d8) {
      Pack8 pk; pk.u = *(const u16x8*)&Qb[l * 256 + h * 32 + d8 * 8];
#pragma unroll
      for (int j = 0; j < 8; ++j)
        s += (float)pk.b[j] * Ksum[h * 32 + d8 * 8 + j];
    }
    Zb[z] = 1.0f / (s + 1e-6f);
  }
  __syncthreads();

  // ---- Phase 2b: per-head KV = K^T V (M=32,N=32,K=64) -> KVt (transposed) -
  for (int t = wave; t < 16; t += 8) {
    int h = t >> 1, mt = t & 1;
    v8f acc0 = {}, acc1 = {};
#pragma unroll
    for (int kk = 0; kk < 2; ++kk) {
      v16bf a  = load_frag_A(KtMb, 64, h * 32 + mt * 16, kk * 32, lane);
      v16bf b0 = load_frag_B(VtM1, 64, h * 32 + 0,  kk * 32, lane);
      v16bf b1 = load_frag_B(VtM1, 64, h * 32 + 16, kk * 32, lane);
      acc0 = wmma_bf16(a, b0, acc0);
      acc1 = wmma_bf16(a, b1, acc1);
    }
    int dp = mt * 16 + 8 * (lane >> 4);          // d' (row) base
    int d0 = (lane & 15), d1 = d0 + 16;          // d  (col)
    Pack8 p0, p1;
#pragma unroll
    for (int j = 0; j < 8; ++j) {
      p0.b[j] = (bf16)acc0[j];
      p1.b[j] = (bf16)acc1[j];
    }
    *(u16x8*)&KVt[(h * 32 + d0) * 32 + dp] = p0.u;
    *(u16x8*)&KVt[(h * 32 + d1) * 32 + dp] = p1.u;
  }
  __syncthreads();

  // ---- Phase 3: msg = (Q . KV) * Z (per head M=64,N=32,K=32) -> Mb --------
  for (int t = wave; t < 32; t += 8) {
    int h = t >> 2, mt = t & 3;
    v16bf a  = load_frag_A(Qb, 256, mt * 16, h * 32, lane);
    v16bf b0 = load_frag_B(KVt, 32, h * 32 + 0,  0, lane);
    v16bf b1 = load_frag_B(KVt, 32, h * 32 + 16, 0, lane);
    v8f acc0 = {}, acc1 = {};
    acc0 = wmma_bf16(a, b0, acc0);
    acc1 = wmma_bf16(a, b1, acc1);
    int tokb = mt * 16 + 8 * (lane >> 4);
    int c0 = h * 32 + (lane & 15), c1 = c0 + 16;
#pragma unroll
    for (int j = 0; j < 8; ++j) {
      int tok = tokb + j;
      float z = Zb[tok * 8 + h];
      KtMb[tok * 256 + c0] = (bf16)(acc0[j] * z);
      KtMb[tok * 256 + c1] = (bf16)(acc1[j] * z);
    }
  }
  __syncthreads();

  // ---- Phase 4: msg @ Wm^T (M=64, N=256, K=256) -> M1 ---------------------
  for (int t = wave; t < 32; t += 8) {
    int mt = t & 3, np = t >> 2;
    v8f acc0 = {}, acc1 = {};
#pragma unroll
    for (int kk = 0; kk < 8; ++kk) {
      v16bf a  = load_frag_A(KtMb, 256, mt * 16, kk * 32, lane);
      v16bf b0 = load_frag_B(Wmb, 256, (np * 2 + 0) * 16, kk * 32, lane);
      v16bf b1 = load_frag_B(Wmb, 256, (np * 2 + 1) * 16, kk * 32, lane);
      acc0 = wmma_bf16(a, b0, acc0);
      acc1 = wmma_bf16(a, b1, acc1);
    }
    int tokb = mt * 16 + 8 * (lane >> 4);
    int c0 = (np * 2) * 16 + (lane & 15), c1 = c0 + 16;
#pragma unroll
    for (int j = 0; j < 8; ++j) {
      VtM1[(tokb + j) * 256 + c0] = (bf16)acc0[j];
      VtM1[(tokb + j) * 256 + c1] = (bf16)acc1[j];
    }
  }
  __syncthreads();

  // ---- Phase 5: LayerNorm(msg) -> Hcat[:, 256:512] ------------------------
  {
    int rrow = tid >> 2, seg = tid & 3;
    float s = 0.0f, s2 = 0.0f;
#pragma unroll
    for (int c8 = 0; c8 < 8; ++c8)
      sumsq8(&VtM1[rrow * 256 + seg * 64 + c8 * 8], s, s2);
    red[(rrow * 4 + seg) * 2 + 0] = s;
    red[(rrow * 4 + seg) * 2 + 1] = s2;
  }
  __syncthreads();
  {
    int rrow = tid >> 2, seg = tid & 3;
    float s = 0.0f, s2 = 0.0f;
#pragma unroll
    for (int q = 0; q < 4; ++q) {
      s  += red[(rrow * 4 + q) * 2 + 0];
      s2 += red[(rrow * 4 + q) * 2 + 1];
    }
    float mean = s * (1.0f / 256.0f);
    float var  = s2 * (1.0f / 256.0f) - mean * mean;
    float rstd = __frsqrt_rn(var + 1e-5f);
    const float4* g14 = (const float4*)g1;
    const float4* b14 = (const float4*)b1;
    for (int c4 = seg * 16; c4 < seg * 16 + 16; ++c4) {
      Pack4 pin; pin.u = *(const u16x4*)&VtM1[rrow * 256 + c4 * 4];
      float4 gv = g14[c4], bv = b14[c4];
      Pack4 po;
      po.b[0] = (bf16)(((float)pin.b[0] - mean) * rstd * gv.x + bv.x);
      po.b[1] = (bf16)(((float)pin.b[1] - mean) * rstd * gv.y + bv.y);
      po.b[2] = (bf16)(((float)pin.b[2] - mean) * rstd * gv.z + bv.z);
      po.b[3] = (bf16)(((float)pin.b[3] - mean) * rstd * gv.w + bv.w);
      *(u16x4*)&Hcat[rrow * 512 + 256 + c4 * 4] = po.u;
    }
  }
  __syncthreads();

  // ---- Phase 6: MLP1 + ReLU (M=64, N=512, K=512) -> H2 --------------------
  for (int t = wave; t < 64; t += 8) {
    int mt = t & 3, np = t >> 2;           // np 0..15
    v8f acc0 = {}, acc1 = {};
#pragma unroll
    for (int kk = 0; kk < 16; ++kk) {
      v16bf a  = load_frag_A(Hcat, 512, mt * 16, kk * 32, lane);
      v16bf b0 = load_frag_B(W1b, 512, (np * 2 + 0) * 16, kk * 32, lane);
      v16bf b1 = load_frag_B(W1b, 512, (np * 2 + 1) * 16, kk * 32, lane);
      acc0 = wmma_bf16(a, b0, acc0);
      acc1 = wmma_bf16(a, b1, acc1);
    }
    int tokb = mt * 16 + 8 * (lane >> 4);
    int c0 = (np * 2) * 16 + (lane & 15), c1 = c0 + 16;
#pragma unroll
    for (int j = 0; j < 8; ++j) {
      H2[(tokb + j) * 512 + c0] = (bf16)fmaxf(acc0[j], 0.0f);
      H2[(tokb + j) * 512 + c1] = (bf16)fmaxf(acc1[j], 0.0f);
    }
  }
  __syncthreads();

  // ---- Phase 7: MLP2 (M=64, N=256, K=512) -> M1 ---------------------------
  for (int t = wave; t < 32; t += 8) {
    int mt = t & 3, np = t >> 2;           // np 0..7
    v8f acc0 = {}, acc1 = {};
#pragma unroll
    for (int kk = 0; kk < 16; ++kk) {
      v16bf a  = load_frag_A(H2, 512, mt * 16, kk * 32, lane);
      v16bf b0 = load_frag_B(W2b, 512, (np * 2 + 0) * 16, kk * 32, lane);
      v16bf b1 = load_frag_B(W2b, 512, (np * 2 + 1) * 16, kk * 32, lane);
      acc0 = wmma_bf16(a, b0, acc0);
      acc1 = wmma_bf16(a, b1, acc1);
    }
    int tokb = mt * 16 + 8 * (lane >> 4);
    int c0 = (np * 2) * 16 + (lane & 15), c1 = c0 + 16;
#pragma unroll
    for (int j = 0; j < 8; ++j) {
      VtM1[(tokb + j) * 256 + c0] = (bf16)acc0[j];
      VtM1[(tokb + j) * 256 + c1] = (bf16)acc1[j];
    }
  }
  __syncthreads();

  // ---- Phase 8: LayerNorm + residual + window-merge scatter ---------------
  {
    int rrow = tid >> 2, seg = tid & 3;
    float s = 0.0f, s2 = 0.0f;
#pragma unroll
    for (int c8 = 0; c8 < 8; ++c8)
      sumsq8(&VtM1[rrow * 256 + seg * 64 + c8 * 8], s, s2);
    red[(rrow * 4 + seg) * 2 + 0] = s;
    red[(rrow * 4 + seg) * 2 + 1] = s2;
  }
  __syncthreads();
  {
    int rrow = tid >> 2, seg = tid & 3;
    float s = 0.0f, s2 = 0.0f;
#pragma unroll
    for (int q = 0; q < 4; ++q) {
      s  += red[(rrow * 4 + q) * 2 + 0];
      s2 += red[(rrow * 4 + q) * 2 + 1];
    }
    float mean = s * (1.0f / 256.0f);
    float var  = s2 * (1.0f / 256.0f) - mean * mean;
    float rstd = __frsqrt_rn(var + 1e-5f);
    int row = wh * 8 + (rrow >> 3);
    int col = ww * 8 + (rrow & 7);
    long base4 = ((long)(b * (HW * HW) + row * HW + col)) * 64;  // float4 units
    const float4* x4   = (const float4*)x;
    float4*       out4 = (float4*)out;
    const float4* g24  = (const float4*)g2;
    const float4* b24  = (const float4*)b2;
    for (int c4 = seg * 16; c4 < seg * 16 + 16; ++c4) {
      Pack4 pin; pin.u = *(const u16x4*)&VtM1[rrow * 256 + c4 * 4];
      float4 xg = x4[base4 + c4];
      float4 gv = g24[c4], bv = b24[c4];
      float4 o;
      o.x = xg.x + ((float)pin.b[0] - mean) * rstd * gv.x + bv.x;
      o.y = xg.y + ((float)pin.b[1] - mean) * rstd * gv.y + bv.y;
      o.z = xg.z + ((float)pin.b[2] - mean) * rstd * gv.z + bv.z;
      o.w = xg.w + ((float)pin.b[3] - mean) * rstd * gv.w + bv.w;
      out4[base4 + c4] = o;
    }
  }
}

extern "C" void kernel_launch(void* const* d_in, const int* in_sizes, int n_in,
                              void* d_out, int out_size, void* d_ws, size_t ws_size,
                              hipStream_t stream) {
  const float* x   = (const float*)d_in[0];
  const float* Wq  = (const float*)d_in[1];
  const float* Wk  = (const float*)d_in[2];
  const float* Wv  = (const float*)d_in[3];
  const float* Wm  = (const float*)d_in[4];
  const float* W1  = (const float*)d_in[5];
  const float* W2  = (const float*)d_in[6];
  const float* g1  = (const float*)d_in[7];
  const float* b1  = (const float*)d_in[8];
  const float* g2  = (const float*)d_in[9];
  const float* b2  = (const float*)d_in[10];
  float* out = (float*)d_out;
  bf16*  wbf = (bf16*)d_ws;

  (void)in_sizes; (void)n_in; (void)out_size; (void)ws_size;

  static_assert(SMEM_BYTES <= 320 * 1024, "fits WGP LDS");
  hipFuncSetAttribute((const void*)loftr_window_kernel,
                      hipFuncAttributeMaxDynamicSharedMemorySize, SMEM_BYTES);

  convert_weights_kernel<<<2560, 256, 0, stream>>>(Wq, Wk, Wv, Wm, W1, W2, wbf);
  loftr_window_kernel<<<NWIN, 256, SMEM_BYTES, stream>>>(x, wbf, g1, b1, g2, b2, out);
}
